// RG_SA3D_9964324126870
// MI455X (gfx1250) — matrix operand order repaired and checked by reading, not compiled, not run on hardware
//
#include <hip/hip_runtime.h>
#include <hip/hip_bf16.h>
#include <math.h>

// ---------------- types ----------------
typedef __bf16 bf16_t;
typedef __attribute__((ext_vector_type(16))) __bf16 v16bf;
typedef __attribute__((ext_vector_type(8)))  float  v8f;
typedef __attribute__((ext_vector_type(4)))  float  v4f;

union FragBF { v16bf v; v4f f4[2]; };

#define NTOK   110592
#define CDIM   384
#define CRDIM  192
#define NHEADS 12
#define MRED   27

// A-fragment (16x32 bf16, M x K) from row-major LDS: lane L holds row L&15;
// chunk0: K = k0+16g .. +7  -> VGPR0..3 ; chunk1: K = k0+16g+8 .. +7 -> VGPR4..7
__device__ __forceinline__ v16bf load_a_lds(const bf16_t* rowptr, int k0, int g) {
  FragBF fr;
  fr.f4[0] = *reinterpret_cast<const v4f*>(rowptr + k0 + 16 * g);
  fr.f4[1] = *reinterpret_cast<const v4f*>(rowptr + k0 + 16 * g + 8);
  return fr.v;
}

// B-fragment (32x16 bf16, K x N) pre-swizzled in global: 32B contiguous per lane
__device__ __forceinline__ v16bf load_b_glb(const bf16_t* base, int fragIdx, int lane) {
  const v4f* p = reinterpret_cast<const v4f*>(base + ((size_t)fragIdx * 32 + (size_t)lane) * 16);
  FragBF fr;
  fr.f4[0] = p[0];
  fr.f4[1] = p[1];
  return fr.v;
}

__device__ __forceinline__ v8f wmma_bf16(v16bf a, v16bf b, v8f c) {
  return __builtin_amdgcn_wmma_f32_16x16x32_bf16(false, a, false, b, (short)0, c, false, false);
}

__device__ __forceinline__ v8f v8f_zero() {
  v8f z = {0.f, 0.f, 0.f, 0.f, 0.f, 0.f, 0.f, 0.f};
  return z;
}

// 64x384 (A in LDS, row-major bf16) x 384x(3*16) (B fragments in global),
// wave computes N-tiles ntBase..ntBase+2 for all 4 M-tiles.
// B double-buffered across the unrolled K loop.
__device__ __forceinline__ void gemm64x3(const bf16_t* __restrict__ Asrc,
                                         const bf16_t* __restrict__ Bfrag,
                                         int ntBase, int ntStride,
                                         int lane, int g, int nloc,
                                         v8f (&acc)[4][3]) {
  FragBF bc[3], bn[3];
#pragma unroll
  for (int j = 0; j < 3; ++j) bc[j].v = load_b_glb(Bfrag, ntBase + j, lane);
#pragma unroll
  for (int kt = 0; kt < 12; ++kt) {
    if (kt < 11) {
#pragma unroll
      for (int j = 0; j < 3; ++j)
        bn[j].v = load_b_glb(Bfrag, (kt + 1) * ntStride + ntBase + j, lane);
    }
    v16bf a[4];
#pragma unroll
    for (int mt = 0; mt < 4; ++mt)
      a[mt] = load_a_lds(Asrc + (mt * 16 + nloc) * CDIM, kt * 32, g);
#pragma unroll
    for (int mt = 0; mt < 4; ++mt)
#pragma unroll
      for (int j = 0; j < 3; ++j)
        acc[mt][j] = wmma_bf16(a[mt], bc[j].v, acc[mt][j]);
    if (kt < 11) {
#pragma unroll
      for (int j = 0; j < 3; ++j) bc[j] = bn[j];
    }
  }
}

// ---------------- stage 1: depthwise 4x4x4 stride-4 (48^3 -> 12^3), float4 ----------------
__global__ __launch_bounds__(256) void k_reduce1(const float* __restrict__ x,
                                                 const float* __restrict__ red_w,
                                                 const float* __restrict__ red_b,
                                                 float* __restrict__ mid) {
  int idx = blockIdx.x * 256 + threadIdx.x;   // 1728 * 96 = 165888 exact
  int c4 = (idx % 96) * 4;
  int p = idx / 96;
  int pz = p / 144, py = (p / 12) % 12, px = p % 12;
  float accv[4];
#pragma unroll
  for (int q = 0; q < 4; ++q) accv[q] = red_b[c4 + q];
#pragma unroll
  for (int a = 0; a < 4; ++a)
#pragma unroll
    for (int b = 0; b < 4; ++b)
#pragma unroll
      for (int e = 0; e < 4; ++e) {
        size_t n = (size_t)((pz * 4 + a) * 48 + (py * 4 + b)) * 48 + (size_t)(px * 4 + e);
        v4f xv = *reinterpret_cast<const v4f*>(x + n * CDIM + c4);
        int widx = a * 16 + b * 4 + e;
#pragma unroll
        for (int q = 0; q < 4; ++q)
          accv[q] = fmaf(red_w[(c4 + q) * 64 + widx], xv[q], accv[q]);
      }
  v4f o = {accv[0], accv[1], accv[2], accv[3]};
  *reinterpret_cast<v4f*>(mid + (size_t)idx * 4) = o;
}

// ---------------- stage 2: depthwise 4x4x4 stride-4 (12^3 -> 3^3) ----------------
__global__ __launch_bounds__(256) void k_red2(const float* __restrict__ mid,
                                              const float* __restrict__ red_w,
                                              const float* __restrict__ red_b,
                                              float* __restrict__ red2) {
  int idx = blockIdx.x * 256 + threadIdx.x;
  if (idx >= MRED * CDIM) return;
  int c = idx % CDIM;
  int p = idx / CDIM;
  int pz = p / 9, py = (p / 3) % 3, px = p % 3;
  const float* w = red_w + c * 64;
  float acc = red_b[c];
#pragma unroll
  for (int a = 0; a < 4; ++a)
#pragma unroll
    for (int b = 0; b < 4; ++b)
#pragma unroll
      for (int e = 0; e < 4; ++e) {
        int q = ((pz * 4 + a) * 12 + (py * 4 + b)) * 12 + (px * 4 + e);
        acc = fmaf(w[a * 16 + b * 4 + e], mid[(size_t)q * CDIM + c], acc);
      }
  red2[idx] = acc;
}

// ---------------- stage 3: depthwise 3x3x3 stride-1 pad-1 on 3^3 ----------------
__global__ __launch_bounds__(256) void k_dw(const float* __restrict__ red2,
                                            const float* __restrict__ dw_w,
                                            const float* __restrict__ dw_b,
                                            float* __restrict__ y) {
  int idx = blockIdx.x * 256 + threadIdx.x;
  if (idx >= MRED * CDIM) return;
  int c = idx % CDIM;
  int p = idx / CDIM;
  int pz = p / 9, py = (p / 3) % 3, px = p % 3;
  float acc = dw_b[c];
#pragma unroll
  for (int a = -1; a <= 1; ++a)
#pragma unroll
    for (int b = -1; b <= 1; ++b)
#pragma unroll
      for (int e = -1; e <= 1; ++e) {
        int z = pz + a, yy = py + b, xx = px + e;
        if (z >= 0 && z < 3 && yy >= 0 && yy < 3 && xx >= 0 && xx < 3)
          acc = fmaf(dw_w[c * 27 + (a + 1) * 9 + (b + 1) * 3 + (e + 1)],
                     red2[((z * 3 + yy) * 3 + xx) * CDIM + c], acc);
      }
  y[idx] = acc;
}

// ---------------- stage 4: 1x1x1 conv + LayerNorm + exact GELU ----------------
__global__ __launch_bounds__(192) void k_conv1ln(const float* __restrict__ y,
                                                 const float* __restrict__ conv_w,
                                                 const float* __restrict__ conv_b,
                                                 const float* __restrict__ ln_g,
                                                 const float* __restrict__ ln_b,
                                                 float* __restrict__ xr) {
  __shared__ float sy[CDIM];
  __shared__ float sz[CRDIM];
  __shared__ float s_mu, s_rstd;
  int p = blockIdx.x;            // 0..26
  int o = threadIdx.x;           // 0..191
  for (int i = o; i < CDIM; i += 192) sy[i] = y[p * CDIM + i];
  __syncthreads();
  float acc = conv_b[o];
  for (int i = 0; i < CDIM; ++i) acc = fmaf(sy[i], conv_w[o * CDIM + i], acc);
  sz[o] = acc;
  __syncthreads();
  if (o == 0) {
    float mu = 0.f;
    for (int i = 0; i < CRDIM; ++i) mu += sz[i];
    mu *= (1.0f / CRDIM);
    float var = 0.f;
    for (int i = 0; i < CRDIM; ++i) { float d = sz[i] - mu; var = fmaf(d, d, var); }
    var *= (1.0f / CRDIM);
    s_mu = mu;
    s_rstd = rsqrtf(var + 1e-5f);
  }
  __syncthreads();
  float u = (sz[o] - s_mu) * s_rstd * ln_g[o] + ln_b[o];
  xr[p * CRDIM + o] = 0.5f * u * (1.0f + erff(u * 0.70710678118654752f));
}

// ---------------- stage 5: K = xr@k_w^T , V = xr@v_w^T ----------------
__global__ __launch_bounds__(256) void k_kv(const float* __restrict__ xr,
                                            const float* __restrict__ k_w,
                                            const float* __restrict__ v_w,
                                            float* __restrict__ Ks_g,
                                            float* __restrict__ Vbuf) {
  int idx = blockIdx.x * 256 + threadIdx.x;
  if (idx < MRED * CRDIM) {
    int o = idx % CRDIM, p = idx / CRDIM;
    float acc = 0.f;
    for (int j = 0; j < CRDIM; ++j) acc = fmaf(xr[p * CRDIM + j], k_w[o * CRDIM + j], acc);
    Ks_g[idx] = acc;
  } else if (idx < MRED * CRDIM + MRED * CDIM) {
    int i2 = idx - MRED * CRDIM;
    int o = i2 % CDIM, p = i2 / CDIM;
    float acc = 0.f;
    for (int j = 0; j < CRDIM; ++j) acc = fmaf(xr[p * CRDIM + j], v_w[o * CRDIM + j], acc);
    Vbuf[i2] = acc;
  }
}

// ---------------- stage 6: cpe(V)+V -> WMMA B-fragment layout (zero-padded m>=27) ----------------
__global__ __launch_bounds__(256) void k_cpe(const float* __restrict__ Vbuf,
                                             const float* __restrict__ cpe_w,
                                             const float* __restrict__ cpe_b,
                                             bf16_t* __restrict__ vwf) {
  int idx = blockIdx.x * 256 + threadIdx.x;  // 24 frags * 32 lanes * 16 = 24576 exact
  int e = idx & 15;
  int L = (idx >> 4) & 31;
  int f = idx >> 9;                 // h*2+nt
  int g = L >> 4, nloc = L & 15;
  int m = g * 16 + e;               // B layout: K = 16*lanegroup + element
  int h = f >> 1, nt = f & 1;
  int col = h * 32 + nt * 16 + nloc;
  float val = 0.f;
  if (m < MRED) {
    int pz = m / 9, py = (m / 3) % 3, px = m % 3;
    float acc = cpe_b[col];
#pragma unroll
    for (int a = -1; a <= 1; ++a)
#pragma unroll
      for (int b = -1; b <= 1; ++b)
#pragma unroll
        for (int e2 = -1; e2 <= 1; ++e2) {
          int z = pz + a, yy = py + b, xx = px + e2;
          if (z >= 0 && z < 3 && yy >= 0 && yy < 3 && xx >= 0 && xx < 3)
            acc = fmaf(cpe_w[col * 27 + (a + 1) * 9 + (b + 1) * 3 + (e2 + 1)],
                       Vbuf[((z * 3 + yy) * 3 + xx) * CDIM + col], acc);
        }
    val = Vbuf[m * CDIM + col] + acc;
  }
  vwf[idx] = (bf16_t)val;
}

// ---------------- stage 7: swizzle q_w / proj_w into bf16 B-fragments ----------------
__global__ __launch_bounds__(256) void k_prepw(const float* __restrict__ qw,
                                               const float* __restrict__ pw,
                                               bf16_t* __restrict__ qwf,
                                               bf16_t* __restrict__ pwf) {
  int idx = blockIdx.x * 256 + threadIdx.x;  // 73728 + 147456 = 221184 exact
  if (idx < 73728) {                          // q_w: 12 kt x 12 nt
    int e = idx & 15, L = (idx >> 4) & 31, fidx = idx >> 9;
    int nt = fidx % 12, kt = fidx / 12;
    int k = kt * 32 + (L >> 4) * 16 + e;
    int n = nt * 16 + (L & 15);
    qwf[idx] = (bf16_t)qw[n * CDIM + k];
  } else {                                    // proj_w: 12 kt x 24 nt
    int j = idx - 73728;
    int e = j & 15, L = (j >> 4) & 31, fidx = j >> 9;
    int nt = fidx % 24, kt = fidx / 24;
    int k = kt * 32 + (L >> 4) * 16 + e;
    int n = nt * 16 + (L & 15);
    pwf[j] = (bf16_t)pw[n * CDIM + k];
  }
}

// ---------------- main fused kernel ----------------
// Block = 128 threads (4 waves) owns 64 tokens. A tile (64x384) shared in LDS;
// N dimension split across waves (wave owns 3 N-tiles) so every weight fragment
// is loaded from L2 exactly once per 64 tokens.
__global__ __launch_bounds__(128) void k_main(const float* __restrict__ x,
                                              const float* __restrict__ Ks_g,
                                              const bf16_t* __restrict__ qwf,
                                              const bf16_t* __restrict__ pwf,
                                              const bf16_t* __restrict__ vwf,
                                              const float* __restrict__ proj_b,
                                              float* __restrict__ out) {
  extern __shared__ char smem[];
  float*  Ks    = reinterpret_cast<float*>(smem);                 // 27*192 f32   (20736 B)
  bf16_t* xa    = reinterpret_cast<bf16_t*>(smem + MRED * CRDIM * 4); // 64*384 bf16 (49152 B), reused as o tile
  bf16_t* qs    = xa + 64 * CDIM;                                 // 64*192 bf16  (24576 B)
  bf16_t* probs = qs + 64 * CRDIM;                                // 12*64*32 bf16(49152 B)

  const int tid = threadIdx.x;
  const int wave = tid >> 5;
  const int lane = tid & 31;
  const int g = lane >> 4, nloc = lane & 15;
  const size_t tok0 = (size_t)blockIdx.x * 64;

  for (int i = tid; i < MRED * CRDIM; i += 128) Ks[i] = Ks_g[i];
  for (int i = tid; i < 64 * CDIM; i += 128) xa[i] = (bf16_t)x[tok0 * CDIM + i];
  __syncthreads();

  // ---- Q = X @ q_w^T : wave owns N-tiles wave*3 .. wave*3+2 over all 64 tokens ----
  {
    v8f acc[4][3];
#pragma unroll
    for (int mt = 0; mt < 4; ++mt)
#pragma unroll
      for (int j = 0; j < 3; ++j) acc[mt][j] = v8f_zero();
    gemm64x3(xa, qwf, wave * 3, 12, lane, g, nloc, acc);
#pragma unroll
    for (int mt = 0; mt < 4; ++mt)
#pragma unroll
      for (int j = 0; j < 3; ++j)
#pragma unroll
        for (int r = 0; r < 8; ++r)
          qs[(mt * 16 + r + 8 * g) * CRDIM + (wave * 3 + j) * 16 + nloc] = (bf16_t)acc[mt][j][r];
  }
  __syncthreads();

  // ---- logits (f32) + exact softmax over 27 keys; scale = (32*0.5)^-0.5 = 0.25 ----
  for (int i = 0; i < 6; ++i) {
    int task = tid + 128 * i;          // 64 tokens x 12 heads = 768 tasks
    int t = task & 63, h = task >> 6;
    float qv[16];
#pragma unroll
    for (int j = 0; j < 16; ++j) qv[j] = (float)qs[t * CRDIM + h * 16 + j];
    const float* krow = Ks + h * 16;
    float l[MRED];
    float mx = -1e30f;
#pragma unroll
    for (int m = 0; m < MRED; ++m) {
      float s = 0.f;
#pragma unroll
      for (int j = 0; j < 16; ++j) s = fmaf(qv[j], krow[m * CRDIM + j], s);
      s *= 0.25f;
      l[m] = s;
      mx = fmaxf(mx, s);
    }
    float sum = 0.f;
#pragma unroll
    for (int m = 0; m < MRED; ++m) { l[m] = expf(l[m] - mx); sum += l[m]; }
    float inv = 1.f / sum;
    bf16_t* prow = probs + (h * 64 + t) * 32;
#pragma unroll
    for (int m = 0; m < MRED; ++m) prow[m] = (bf16_t)(l[m] * inv);
#pragma unroll
    for (int m = MRED; m < 32; ++m) prow[m] = (bf16_t)0.f;
  }
  __syncthreads();

  // ---- O = attn @ V' : wave owns its 16 tokens (rows wave*16..), all heads ----
  bf16_t* os = xa;   // reuse x tile storage for o tile (all xa reads finished pre-barrier)
  for (int h = 0; h < NHEADS; ++h) {
    v16bf a = load_a_lds(probs + (h * 64 + wave * 16 + nloc) * 32, 0, g);
#pragma unroll
    for (int nt = 0; nt < 2; ++nt) {
      v16bf b = load_b_glb(vwf, h * 2 + nt, lane);
      v8f acc = v8f_zero();
      acc = wmma_bf16(a, b, acc);
#pragma unroll
      for (int r = 0; r < 8; ++r)
        os[(wave * 16 + r + 8 * g) * CDIM + h * 32 + nt * 16 + nloc] = (bf16_t)acc[r];
    }
  }
  __syncthreads();

  // ---- OUT = O @ proj_w^T + proj_b : two 192-col halves, wave owns 3 N-tiles each ----
  for (int half = 0; half < 2; ++half) {
    v8f acc[4][3];
#pragma unroll
    for (int mt = 0; mt < 4; ++mt)
#pragma unroll
      for (int j = 0; j < 3; ++j) acc[mt][j] = v8f_zero();
    gemm64x3(os, pwf, half * 12 + wave * 3, 24, lane, g, nloc, acc);
#pragma unroll
    for (int mt = 0; mt < 4; ++mt)
#pragma unroll
      for (int j = 0; j < 3; ++j) {
        int col = (half * 12 + wave * 3 + j) * 16 + nloc;
        float bias = proj_b[col];
#pragma unroll
        for (int r = 0; r < 8; ++r)
          out[(tok0 + (size_t)(mt * 16 + r + 8 * g)) * CDIM + col] = acc[mt][j][r] + bias;
      }
  }
}

// ---------------- launch ----------------
extern "C" void kernel_launch(void* const* d_in, const int* in_sizes, int n_in,
                              void* d_out, int out_size, void* d_ws, size_t ws_size,
                              hipStream_t stream) {
  (void)in_sizes; (void)n_in; (void)out_size; (void)ws_size;
  const float* x      = (const float*)d_in[0];
  const float* red_w  = (const float*)d_in[1];
  const float* red_b  = (const float*)d_in[2];
  const float* dw_w   = (const float*)d_in[3];
  const float* dw_b   = (const float*)d_in[4];
  const float* conv_w = (const float*)d_in[5];
  const float* conv_b = (const float*)d_in[6];
  const float* ln_g   = (const float*)d_in[7];
  const float* ln_b   = (const float*)d_in[8];
  const float* q_w    = (const float*)d_in[9];
  const float* k_w    = (const float*)d_in[10];
  const float* v_w    = (const float*)d_in[11];
  const float* cpe_w  = (const float*)d_in[12];
  const float* cpe_b  = (const float*)d_in[13];
  const float* proj_w = (const float*)d_in[14];
  const float* proj_b = (const float*)d_in[15];
  float* out = (float*)d_out;

  char* ws = (char*)d_ws;
  float*  mid  = (float*)(ws + 0);             // 1728*384 f32      = 2,654,208 B
  float*  red2 = (float*)(ws + 2654208);       // 27*384 f32        = 41,472
  float*  ybuf = (float*)(ws + 2695680);       // 27*384 f32        = 41,472
  float*  xr   = (float*)(ws + 2737152);       // 27*192 f32        = 20,736
  float*  Vbuf = (float*)(ws + 2757888);       // 27*384 f32        = 41,472
  float*  Ks_g = (float*)(ws + 2799360);       // 27*192 f32        = 20,736
  bf16_t* vwf  = (bf16_t*)(ws + 2820096);      // 24*512 bf16       = 24,576
  bf16_t* qwf  = (bf16_t*)(ws + 2844672);      // 144*512 bf16      = 147,456
  bf16_t* pwf  = (bf16_t*)(ws + 2992128);      // 288*512 bf16      = 294,912

  k_reduce1<<<648,  256, 0, stream>>>(x, red_w, red_b, mid);
  k_red2   <<<41,   256, 0, stream>>>(mid, red_w, red_b, red2);
  k_dw     <<<41,   256, 0, stream>>>(red2, dw_w, dw_b, ybuf);
  k_conv1ln<<<27,   192, 0, stream>>>(ybuf, conv_w, conv_b, ln_g, ln_b, xr);
  k_kv     <<<61,   256, 0, stream>>>(xr, k_w, v_w, Ks_g, Vbuf);
  k_cpe    <<<96,   256, 0, stream>>>(Vbuf, cpe_w, cpe_b, vwf);
  k_prepw  <<<864,  256, 0, stream>>>(q_w, proj_w, qwf, pwf);

  // dynamic LDS: 27*192*4 + 64*384*2 + 64*192*2 + 12*64*32*2 = 143,616 B
  size_t lds = (size_t)(MRED * CRDIM * 4) + 64 * CDIM * 2 + 64 * CRDIM * 2 + NHEADS * 64 * 32 * 2;
  k_main   <<<1728, 128, lds, stream>>>(x, Ks_g, qwf, pwf, vwf, proj_b, out);
}
